// GroupedQueryAttention_65008624992775
// MI455X (gfx1250) — matrix operand order repaired and checked by reading, not compiled, not run on hardware
//
#include <hip/hip_runtime.h>
#include <hip/hip_bf16.h>

// MI455X (gfx1250) implementation.
// - bf16 WMMA (v_wmma_f32_16x16x32_bf16) for the four big GEMMs (compute-bound),
//   fp32 accumulation.
// - GEMM tiles staged with the Tensor Data Mover (TENSOR_LOAD_TO_LDS), double
//   buffered in LDS, TDM descriptor padding gives an 80B row pitch (bank-friendly
//   16B-aligned ds_load_b128 fragment reads).
// - one-wave-per-token WMMA attention for the per-token 16x16 head-attention.
// Workspace usage: ~352 MiB.

typedef __bf16 bf16_t;
typedef __attribute__((ext_vector_type(16))) __bf16 v16bf;
typedef __attribute__((ext_vector_type(8)))  __bf16 v8bf;
typedef __attribute__((ext_vector_type(4)))  __bf16 v4bf;
typedef __attribute__((ext_vector_type(8)))  float  v8f;
typedef __attribute__((ext_vector_type(4)))  unsigned int v4u;
typedef __attribute__((ext_vector_type(8)))  int v8i;
typedef __attribute__((ext_vector_type(4)))  int v4i;

#define WMMA_BF16(a, b, c) \
  __builtin_amdgcn_wmma_f32_16x16x32_bf16(false, (a), false, (b), (short)0, (c), false, false)

static __device__ __forceinline__ v16bf cat8(v8bf lo, v8bf hi) {
  return __builtin_shufflevector(lo, hi, 0,1,2,3,4,5,6,7,8,9,10,11,12,13,14,15);
}

// ---------------------------------------------------------------------------
// Tensor Data Mover: 2D tile (tile_k_elems x tile_rows) of bf16 from global
// (row pitch stride_elems) into LDS at lds_off, with 16B padding inserted
// after every 64B row -> 80B LDS row pitch.
// D# packing per CDNA5 ISA 8.3/8.4; groups 2/3 zero (2D tensor).
// Toolchain uses the 6-arg builtin: (g0 v4u, g1 v8i, g2 v4i, g3 v4i, v8i, cpol).
// ---------------------------------------------------------------------------
static __device__ __forceinline__ void tdm_load_tile(unsigned lds_off,
                                                     const void* gsrc,
                                                     unsigned tile_k_elems,
                                                     unsigned tile_rows,
                                                     unsigned stride_elems) {
  unsigned long long ga = (unsigned long long)(size_t)gsrc;
  v4u g0;
  g0[0] = 1u;                                  // count=1, user descriptor
  g0[1] = lds_off;                             // lds_addr (bytes)
  g0[2] = (unsigned)ga;                        // global_addr[31:0]
  g0[3] = (unsigned)((ga >> 32) & 0x01ffffffu) | (2u << 30);  // [56:32] | type=2

  const unsigned td0 = 0x40000000u;  // huge tensor dims: tiles always in-bounds
  const unsigned td1 = 0x40000000u;
  v8i g1;
  g1[0] = (int)((1u << 16)     // data_size = 2 bytes
              | (1u << 20)     // pad_enable
              | (3u << 22)     // pad_interval: every 16 DWORDs (64B row)
              | (3u << 25));   // pad_amount: 4 DWORDs (16B) -> 80B pitch
  g1[1] = (int)((td0 & 0xffffu) << 16);                  // tensor_dim0[15:0]
  g1[2] = (int)((td0 >> 16) | ((td1 & 0xffffu) << 16));  // dim0 hi | dim1 lo
  g1[3] = (int)((td1 >> 16) | (tile_k_elems << 16));     // dim1 hi | tile_dim0
  g1[4] = (int)(tile_rows & 0xffffu);                    // tile_dim1 (tile_dim2=0)
  g1[5] = (int)stride_elems;                             // tensor_dim0_stride lo32
  g1[6] = 0;                                             // stride hi | dim1_stride
  g1[7] = 0;
  v4i gz4 = {0, 0, 0, 0};
  v8i gz8 = {0, 0, 0, 0, 0, 0, 0, 0};
  __builtin_amdgcn_tensor_load_to_lds(g0, g1, gz4, gz4, gz8, 0);
}

static __device__ __forceinline__ unsigned lds_offset_of(const void* p) {
  // generic LDS address: aperture in [63:32], LDS byte offset in [31:0]
  return (unsigned)(size_t)p;
}

// ---------------------------------------------------------------------------
// fp32 -> bf16 straight conversion (vectorized)
// ---------------------------------------------------------------------------
__global__ __launch_bounds__(256)
void cvt_f32_bf16(const float* __restrict__ in, bf16_t* __restrict__ out, long n) {
  long i = ((long)blockIdx.x * blockDim.x + threadIdx.x) * 4;
  if (i + 3 < n) {
    float4 f = *(const float4*)(in + i);
    v4bf o;
    o[0] = (bf16_t)f.x; o[1] = (bf16_t)f.y; o[2] = (bf16_t)f.z; o[3] = (bf16_t)f.w;
    *(v4bf*)(out + i) = o;
  }
}

// ---------------------------------------------------------------------------
// fp32 [K,N] -> bf16 [N,K] tiled transpose+convert (coalesced via LDS)
// ---------------------------------------------------------------------------
__global__ __launch_bounds__(256)
void transpose_cvt(const float* __restrict__ W, bf16_t* __restrict__ Wt, int D) {
  __shared__ float t[32][33];
  const int bx = blockIdx.x * 32, by = blockIdx.y * 32;
  const int tx = threadIdx.x & 31, ty = threadIdx.x >> 5;  // 32 x 8
  #pragma unroll
  for (int i = 0; i < 32; i += 8)
    t[ty + i][tx] = W[(long)(by + ty + i) * D + bx + tx];
  __syncthreads();
  #pragma unroll
  for (int i = 0; i < 32; i += 8)
    Wt[(long)(bx + ty + i) * D + by + tx] = (bf16_t)t[tx][ty + i];
}

// ---------------------------------------------------------------------------
// C[M,N] = A[M,K](bf16) * Bt[N,K]^T(bf16) + bias, via WMMA bf16 + TDM staging.
// Block tile 128x128, 8 waves, each wave 32(M) x 64(N) = 2x4 WMMA tiles.
// Double-buffered LDS; wave0 DMAs the A tile, wave1 the B tile (TENSORcnt).
// MODE 0: bf16 row-major out (q,k)
// MODE 1: bf16 out, per-token head-transposed: out[t, dim*16+head] (v)
// MODE 2: fp32 row-major out (final projection)
// ---------------------------------------------------------------------------
template <int MODE>
__global__ __launch_bounds__(256)
void gemm_bf16_wmma(const bf16_t* __restrict__ A, const bf16_t* __restrict__ Bt,
                    const float* __restrict__ bias, void* __restrict__ Cout,
                    int M, int N, int K) {
  // 40-elem (80B) row pitch produced by TDM pad fields; 16B aligned rows
  __shared__ __align__(16) bf16_t sA[2][128][40];
  __shared__ __align__(16) bf16_t sB[2][128][40];

  const int tid  = threadIdx.x;
  const int lane = tid & 31;
  const int wid  = tid >> 5;
  const int wm   = (wid & 3) * 32;   // wave M offset in block tile
  const int wn   = (wid >> 2) * 64;  // wave N offset in block tile
  const int lrow = lane & 15;
  const bool hih = lane >= 16;
  const int m0 = blockIdx.y * 128;
  const int n0 = blockIdx.x * 128;

  v8f zero = {};
  v8f acc[2][4];
  #pragma unroll
  for (int i = 0; i < 2; i++)
    #pragma unroll
    for (int j = 0; j < 4; j++) acc[i][j] = zero;

  const int kA = hih ? 16 : 0;   // A-operand: lanes>=16 take K 16..31
  const int kB = hih ? 16 : 0;   // B-operand: lanes>=16 take K 16..31

  const int nk = K / 32;
  // prologue: DMA tile 0 into buffer 0
  if (wid == 0)
    tdm_load_tile(lds_offset_of(&sA[0][0][0]), A + (long)m0 * K, 32, 128, K);
  else if (wid == 1)
    tdm_load_tile(lds_offset_of(&sB[0][0][0]), Bt + (long)n0 * K, 32, 128, K);

  for (int kt = 0; kt < nk; ++kt) {
    const int cur = kt & 1;
    if (kt + 1 < nk) {
      const long koff = (long)(kt + 1) * 32;
      if (wid == 0)
        tdm_load_tile(lds_offset_of(&sA[cur ^ 1][0][0]),
                      A + (long)m0 * K + koff, 32, 128, K);
      else if (wid == 1)
        tdm_load_tile(lds_offset_of(&sB[cur ^ 1][0][0]),
                      Bt + (long)n0 * K + koff, 32, 128, K);
      if (wid < 2) __builtin_amdgcn_s_wait_tensorcnt(1);  // tile kt complete
    } else {
      if (wid < 2) __builtin_amdgcn_s_wait_tensorcnt(0);
    }
    __syncthreads();  // publish buf[cur]

    v16bf af[2], bf[4];
    #pragma unroll
    for (int i = 0; i < 2; i++) {
      v8bf lo = *(const v8bf*)&sA[cur][wm + i * 16 + lrow][kA];
      v8bf hi = *(const v8bf*)&sA[cur][wm + i * 16 + lrow][kA + 8];
      af[i] = cat8(lo, hi);
    }
    #pragma unroll
    for (int j = 0; j < 4; j++) {
      v8bf lo = *(const v8bf*)&sB[cur][wn + j * 16 + lrow][kB];
      v8bf hi = *(const v8bf*)&sB[cur][wn + j * 16 + lrow][kB + 8];
      bf[j] = cat8(lo, hi);
    }
    #pragma unroll
    for (int i = 0; i < 2; i++)
      #pragma unroll
      for (int j = 0; j < 4; j++)
        acc[i][j] = WMMA_BF16(af[i], bf[j], acc[i][j]);

    // all LDS reads of buf[cur] retired before next DMA may overwrite it
    asm volatile("s_wait_dscnt 0" ::: "memory");
    __syncthreads();
  }

  // epilogue: C layout -> lanes<16 rows r, lanes>=16 rows r+8; col = lane&15
  #pragma unroll
  for (int i = 0; i < 2; i++) {
    #pragma unroll
    for (int j = 0; j < 4; j++) {
      const int col  = n0 + wn + j * 16 + lrow;
      const float bv = bias[col];
      const int rb   = m0 + wm + i * 16 + (hih ? 8 : 0);
      #pragma unroll
      for (int r = 0; r < 8; r++) {
        const int row = rb + r;
        const float val = acc[i][j][r] + bv;
        if (MODE == 0) {
          ((bf16_t*)Cout)[(long)row * N + col] = (bf16_t)val;
        } else if (MODE == 1) {
          ((bf16_t*)Cout)[(long)row * N + ((col & 127) << 4) + (col >> 7)] = (bf16_t)val;
        } else {
          ((float*)Cout)[(long)row * N + col] = val;
        }
      }
    }
  }
}

// ---------------------------------------------------------------------------
// Per-token head attention: one wave per token.
// scores = (q_perm @ k^T)/sqrt(128)  [16x16], softmax over key-heads,
// attn = P @ v  [16 x 128]. P re-laid-out C->A through LDS, K padded 16->32.
// ---------------------------------------------------------------------------
__global__ __launch_bounds__(256)
void attn_heads(const bf16_t* __restrict__ q, const bf16_t* __restrict__ k,
                const bf16_t* __restrict__ vt, bf16_t* __restrict__ attn) {
  __shared__ __align__(16) bf16_t sP[8][16][32];
  const int tid  = threadIdx.x;
  const int lane = tid & 31;
  const int wid  = tid >> 5;
  const int lrow = lane & 15;
  const bool hih = lane >= 16;
  const long token = (long)blockIdx.x * 8 + wid;

  const bf16_t* qrow = q  + token * 2048;
  const bf16_t* krow = k  + token * 2048;
  const bf16_t* vrow = vt + token * 2048;

  // GQA permutation: logical head m reads stored head (m%4)*4 + m/4  (G=4, H/G=4)
  const int qhead = ((lrow & 3) << 2) | (lrow >> 2);
  const int kA = hih ? 16 : 0;
  const int kB = hih ? 16 : 0;

  v8f zero = {};
  v8f c = zero;
  #pragma unroll
  for (int kk = 0; kk < 4; kk++) {
    const bf16_t* pa = qrow + qhead * 128 + kk * 32;
    v16bf a = cat8(*(const v8bf*)(pa + kA), *(const v8bf*)(pa + kA + 8));
    const bf16_t* pb = krow + lrow * 128 + kk * 32 + kB;
    v16bf b = cat8(*(const v8bf*)(pb), *(const v8bf*)(pb + 8));
    c = WMMA_BF16(a, b, c);
  }

  // softmax over key-heads (N axis = 16 lanes of each half-wave)
  const float scale = 0.08838834764831845f;  // 1/sqrt(128)
  bf16_t pv[8];
  #pragma unroll
  for (int r = 0; r < 8; r++) {
    float s = c[r] * scale;
    float mx = s;
    mx = fmaxf(mx, __shfl_xor(mx, 1));
    mx = fmaxf(mx, __shfl_xor(mx, 2));
    mx = fmaxf(mx, __shfl_xor(mx, 4));
    mx = fmaxf(mx, __shfl_xor(mx, 8));
    float e = __expf(s - mx);
    float sum = e;
    sum += __shfl_xor(sum, 1);
    sum += __shfl_xor(sum, 2);
    sum += __shfl_xor(sum, 4);
    sum += __shfl_xor(sum, 8);
    pv[r] = (bf16_t)(e / sum);
  }

  // C layout -> LDS (rows m, cols = key-head), zero-pad K 16..31
  #pragma unroll
  for (int r = 0; r < 8; r++) {
    const int m = (hih ? 8 : 0) + r;
    sP[wid][m][lrow]      = pv[r];
    sP[wid][m][16 + lrow] = (bf16_t)0.0f;
  }
  __syncthreads();

  // read P back as A-operand (16x32, upper K half zero)
  v16bf pa = cat8(*(const v8bf*)&sP[wid][lrow][kA],
                  *(const v8bf*)&sP[wid][lrow][kA + 8]);

  // attn = P @ V : 8 N-tiles of 16 dims; B-operand from head-transposed v
  #pragma unroll
  for (int j = 0; j < 8; j++) {
    v16bf bv;
    if (!hih) {
      const bf16_t* pb = vrow + (j * 16 + lrow) * 16;  // 16 heads contiguous
      bv = cat8(*(const v8bf*)pb, *(const v8bf*)(pb + 8));
    } else {
      #pragma unroll
      for (int e = 0; e < 16; e++) bv[e] = (bf16_t)0.0f;  // padded K rows 16..31
    }
    v8f o = WMMA_BF16(pa, bv, zero);
    #pragma unroll
    for (int r = 0; r < 8; r++) {
      const int m = (hih ? 8 : 0) + r;
      attn[token * 2048 + m * 128 + j * 16 + lrow] = (bf16_t)o[r];
    }
  }
}

// ---------------------------------------------------------------------------
extern "C" void kernel_launch(void* const* d_in, const int* in_sizes, int n_in,
                              void* d_out, int out_size, void* d_ws, size_t ws_size,
                              hipStream_t stream) {
  (void)in_sizes; (void)n_in; (void)out_size; (void)ws_size;
  const float* x  = (const float*)d_in[0];
  const float* Wq = (const float*)d_in[1];
  const float* bq = (const float*)d_in[2];
  const float* Wk = (const float*)d_in[3];
  const float* bk = (const float*)d_in[4];
  const float* Wv = (const float*)d_in[5];
  const float* bv = (const float*)d_in[6];
  const float* Wo = (const float*)d_in[7];
  const float* bo = (const float*)d_in[8];
  // num_heads=16, num_groups=4 hardcoded per reference setup.

  const int M = 4 * 4096;  // B*S tokens
  const int D = 2048;

  char* p = (char*)d_ws;
  auto carve = [&](size_t bytes) -> char* {
    char* r = p;
    p += (bytes + 255) & ~(size_t)255;
    return r;
  };
  bf16_t* xb  = (bf16_t*)carve((size_t)M * D * 2);
  bf16_t* Wqt = (bf16_t*)carve((size_t)D * D * 2);
  bf16_t* Wkt = (bf16_t*)carve((size_t)D * D * 2);
  bf16_t* Wvt = (bf16_t*)carve((size_t)D * D * 2);
  bf16_t* Wot = (bf16_t*)carve((size_t)D * D * 2);
  bf16_t* qb  = (bf16_t*)carve((size_t)M * D * 2);
  bf16_t* kb  = (bf16_t*)carve((size_t)M * D * 2);
  bf16_t* vtb = (bf16_t*)carve((size_t)M * D * 2);
  bf16_t* ab  = (bf16_t*)carve((size_t)M * D * 2);

  const long nx = (long)M * D;
  cvt_f32_bf16<<<(int)(nx / (256 * 4)), 256, 0, stream>>>(x, xb, nx);

  dim3 tg(D / 32, D / 32);
  transpose_cvt<<<tg, 256, 0, stream>>>(Wq, Wqt, D);
  transpose_cvt<<<tg, 256, 0, stream>>>(Wk, Wkt, D);
  transpose_cvt<<<tg, 256, 0, stream>>>(Wv, Wvt, D);
  transpose_cvt<<<tg, 256, 0, stream>>>(Wo, Wot, D);

  dim3 gg(D / 128, M / 128);
  gemm_bf16_wmma<0><<<gg, 256, 0, stream>>>(xb, Wqt, bq, qb, M, D, D);
  gemm_bf16_wmma<0><<<gg, 256, 0, stream>>>(xb, Wkt, bk, kb, M, D, D);
  gemm_bf16_wmma<1><<<gg, 256, 0, stream>>>(xb, Wvt, bv, vtb, M, D, D);

  attn_heads<<<M / 8, 256, 0, stream>>>(qb, kb, vtb, ab);

  gemm_bf16_wmma<2><<<gg, 256, 0, stream>>>(ab, Wot, bo, d_out, M, D, D);
}
